// AtomAttentionDecoder_37031208026672
// MI455X (gfx1250) — matrix operand order repaired and checked by reading, not compiled, not run on hardware
//
#include <hip/hip_runtime.h>
#include <hip/hip_bf16.h>

typedef __attribute__((ext_vector_type(16))) _Float16 v16h;
typedef __attribute__((ext_vector_type(8)))  _Float16 v8h;
typedef __attribute__((ext_vector_type(8)))  float    v8f;

#define N_TOK   2048
#define N_ATOMS 16384
#define C_ATOM  128
#define C_PAIR  16
#define NQ      32
#define NK      128
#define NHEADS  4
#define NW      (N_ATOMS / NQ)     // 512 windows
#define PAD     ((NK - NQ) / 2)    // 48
#define BATCH   4
#define SCALE_ATTN 0.17677669529663687f  // 1/sqrt(32)

// ---------------------------------------------------------------------------
// Fragment helpers (16x16x32 f16 WMMA, wave32)
// A layout: lane m=l%16; halves e -> K = e + 16*(l>=16) (matches B).
// B layout: lane n=l%16; halves e -> K = e + 16*(l>=16).
// C layout: lane n=l%16; vgpr v -> M = v + 8*(l>=16).
// ---------------------------------------------------------------------------
__device__ __forceinline__ v8f wmma16(v16h a, v16h b, v8f c) {
  return __builtin_amdgcn_wmma_f32_16x16x32_f16(false, a, false, b, (short)0, c,
                                                false, false);
}

__device__ __forceinline__ v16h load_fragA_f16(const _Float16* P, int ld, int r0,
                                               int k0, int lane) {
  const _Float16* p = P + (size_t)(r0 + (lane & 15)) * ld + k0 + ((lane >> 4) << 4);
  v16h f;
#pragma unroll
  for (int e = 0; e < 8; ++e) { f[e] = p[e]; f[8 + e] = p[8 + e]; }
  return f;
}

__device__ __forceinline__ v16h load_fragA_f32(const float* P, int ld, int r0,
                                               int k0, int lane) {
  const float* p = P + (size_t)(r0 + (lane & 15)) * ld + k0 + ((lane >> 4) << 4);
  v16h f;
#pragma unroll
  for (int e = 0; e < 8; ++e) {
    f[e]     = (_Float16)p[e];
    f[8 + e] = (_Float16)p[8 + e];
  }
  return f;
}

__device__ __forceinline__ v16h load_fragB_f16(const _Float16* P, int ld, int n0,
                                               int k0, int lane) {
  const _Float16* p = P + (size_t)(n0 + (lane & 15)) * ld + k0 + ((lane >> 4) << 4);
  v16h f;
#pragma unroll
  for (int e = 0; e < 16; ++e) f[e] = p[e];
  return f;
}

__device__ __forceinline__ v8f v8f_zero() {
  v8f z = {0.f, 0.f, 0.f, 0.f, 0.f, 0.f, 0.f, 0.f};
  return z;
}

// Async global->LDS 16B copy (per-lane), tracked by ASYNCcnt.
__device__ __forceinline__ void async_copy_b128(unsigned lds_off,
                                                const void* gaddr) {
  asm volatile("global_load_async_to_lds_b128 %0, %1, off"
               :: "v"(lds_off), "v"((unsigned long long)gaddr) : "memory");
}
__device__ __forceinline__ void wait_asynccnt0() {
  asm volatile("s_wait_asynccnt 0x0" ::: "memory");
}

// ---------------------------------------------------------------------------
// Weight prep: fp32 [K,N] row-major -> f16 [N,K] row-major (transposed)
// ---------------------------------------------------------------------------
__global__ void k_transpose_f32_f16(const float* __restrict__ src,
                                    _Float16* __restrict__ dst, int K, int N) {
  int i = blockIdx.x * blockDim.x + threadIdx.x;
  if (i >= K * N) return;
  int k = i / N, n = i - k * N;
  dst[(size_t)n * K + k] = (_Float16)src[(size_t)k * N + n];
}

// ---------------------------------------------------------------------------
// Generic WMMA GEMM: C[M,N] = A[M,K] @ BT[N,K]^T
// block = 128 threads (4 waves), tile 64x64; all waves share the B tile,
// which is staged once into LDS with async global->LDS loads.
// OMODE: 0 = store f32, 1 = accumulate into f32, 2 = store f16
// ---------------------------------------------------------------------------
template <bool AF32, int OMODE, int KMAX>
__global__ void k_gemm_wmma(const void* __restrict__ Av,
                            const _Float16* __restrict__ BT, void* __restrict__ Cv,
                            int M, int N, int K) {
  __shared__ __align__(16) _Float16 Bs[64 * KMAX];
  int lane = threadIdx.x & 31;
  int wave = threadIdx.x >> 5;
  int m0 = blockIdx.x * 64 + wave * 16;
  int n0 = blockIdx.y * 64;

  // Stage BT rows [n0, n0+64) (contiguous 64*K halves) into LDS, async.
  {
    const char* src = (const char*)(BT + (size_t)n0 * K);
    unsigned lbase = (unsigned)(unsigned long long)&Bs[0];
    int bytes = 64 * K * 2;
    for (int off = threadIdx.x * 16; off < bytes; off += 128 * 16)
      async_copy_b128(lbase + off, src + off);
    wait_asynccnt0();
  }
  __syncthreads();

  v8f acc[4] = {v8f_zero(), v8f_zero(), v8f_zero(), v8f_zero()};
  for (int k0 = 0; k0 < K; k0 += 32) {
    v16h a = AF32 ? load_fragA_f32((const float*)Av, K, m0, k0, lane)
                  : load_fragA_f16((const _Float16*)Av, K, m0, k0, lane);
#pragma unroll
    for (int nt = 0; nt < 4; ++nt) {
      v16h b = load_fragB_f16(Bs, K, nt * 16, k0, lane);   // LDS broadcast
      acc[nt] = wmma16(a, b, acc[nt]);
    }
  }
  int mrow = m0 + ((lane >> 4) << 3);
#pragma unroll
  for (int nt = 0; nt < 4; ++nt) {
    int n = n0 + nt * 16 + (lane & 15);
    if (OMODE == 2) {
      _Float16* C = (_Float16*)Cv;
#pragma unroll
      for (int vv = 0; vv < 8; ++vv)
        C[(size_t)(mrow + vv) * N + n] = (_Float16)acc[nt][vv];
    } else if (OMODE == 1) {
      float* C = (float*)Cv;
#pragma unroll
      for (int vv = 0; vv < 8; ++vv) C[(size_t)(mrow + vv) * N + n] += acc[nt][vv];
    } else {
      float* C = (float*)Cv;
#pragma unroll
      for (int vv = 0; vv < 8; ++vv) C[(size_t)(mrow + vv) * N + n] = acc[nt][vv];
    }
  }
}

// ---------------------------------------------------------------------------
// Gather: x[b,i,:] = a_tok[b, idx[b,i], :]     grid = B*N_ATOMS, 128 threads
// ---------------------------------------------------------------------------
__global__ void k_gather(const float* __restrict__ a_tok,
                         const int* __restrict__ idx, float* __restrict__ x) {
  size_t row = blockIdx.x;
  int b = (int)(row / N_ATOMS);
  int tok = idx[row];
  x[row * C_ATOM + threadIdx.x] =
      a_tok[((size_t)b * N_TOK + tok) * C_ATOM + threadIdx.x];
}

// ---------------------------------------------------------------------------
// Pair embedding + attention bias (tiny: only first NK atoms/batch matter)
// ---------------------------------------------------------------------------
__global__ void k_pair_bias(const float* __restrict__ x, const float* __restrict__ Wcl,
                            const float* __restrict__ Wcm, const float* __restrict__ Wm1,
                            const float* __restrict__ Wm2, const float* __restrict__ Wpb,
                            float* __restrict__ bias) {
  __shared__ float p[NK][C_PAIR];
  int b = blockIdx.x, t = threadIdx.x;
  const float* xr = x + ((size_t)b * N_ATOMS + t) * C_ATOM;
  float c0[C_PAIR];
#pragma unroll 4
  for (int j = 0; j < C_PAIR; ++j) {
    float s = 0.f;
    for (int k = 0; k < C_ATOM; ++k) s += xr[k] * (Wcl[k * C_PAIR + j] + Wcm[k * C_PAIR + j]);
    c0[j] = s;
  }
  float c1[C_PAIR];
#pragma unroll
  for (int j = 0; j < C_PAIR; ++j) {
    float s = 0.f;
#pragma unroll
    for (int k = 0; k < C_PAIR; ++k) s += fmaxf(c0[k], 0.f) * Wm1[k * C_PAIR + j];
    c1[j] = s;
  }
#pragma unroll
  for (int j = 0; j < C_PAIR; ++j) {
    float s = 0.f;
#pragma unroll
    for (int k = 0; k < C_PAIR; ++k) s += fmaxf(c1[k], 0.f) * Wm2[k * C_PAIR + j];
    p[t][j] = s;
  }
  __syncthreads();
  for (int i = t; i < NHEADS * NQ * NK; i += 128) {
    int h = i / (NQ * NK);
    int q = (i / NK) % NQ;
    int k = i % NK;
    float s = 0.f;
#pragma unroll
    for (int c = 0; c < C_PAIR; ++c) s += (p[q][c] + p[k][c]) * Wpb[c * NHEADS + h];
    bias[(((size_t)b * NHEADS + h) * NQ + q) * NK + k] = s;
  }
}

// ---------------------------------------------------------------------------
// LayerNorm (per 128-elem row) -> f16. 256 threads = 8 waves, wave per row.
// ---------------------------------------------------------------------------
__global__ void k_layernorm_f16(const float* __restrict__ x, const float* __restrict__ g,
                                const float* __restrict__ bb, _Float16* __restrict__ h) {
  int lane = threadIdx.x & 31;
  size_t row = (size_t)blockIdx.x * 8 + (threadIdx.x >> 5);
  const float* xr = x + row * C_ATOM;
  float vals[4], s = 0.f, s2 = 0.f;
#pragma unroll
  for (int j = 0; j < 4; ++j) {
    float v = xr[lane + 32 * j];
    vals[j] = v; s += v; s2 += v * v;
  }
#pragma unroll
  for (int off = 16; off > 0; off >>= 1) {
    s  += __shfl_xor(s, off, 32);
    s2 += __shfl_xor(s2, off, 32);
  }
  float mean = s * (1.f / C_ATOM);
  float var = s2 * (1.f / C_ATOM) - mean * mean;
  float rinv = rsqrtf(var + 1e-5f);
#pragma unroll
  for (int j = 0; j < 4; ++j) {
    int c = lane + 32 * j;
    h[row * C_ATOM + c] = (_Float16)((vals[j] - mean) * rinv * g[c] + bb[c]);
  }
}

// ---------------------------------------------------------------------------
// Windowed attention. grid = B*NW*2 (head pairs), 128 threads = 4 waves.
// Phases: WMMA raw scores -> LDS; softmax row pass applies scale+bias+mask
// (contiguous bias row reads); stage V^T in LDS (reusing score space);
// WMMA P@V. LDS: 48 KB total.
// ---------------------------------------------------------------------------
__device__ __forceinline__ v16h load_fragB_kclamp(const _Float16* Kb, int r0, int d0,
                                                  int lane) {
  int row = r0 + (lane & 15);
  row = row < 0 ? 0 : (row > N_ATOMS - 1 ? N_ATOMS - 1 : row);
  const _Float16* p = Kb + (size_t)row * C_ATOM + d0 + ((lane >> 4) << 4);
  v16h f;
#pragma unroll
  for (int e = 0; e < 16; ++e) f[e] = p[e];
  return f;
}

__global__ void k_attention(const _Float16* __restrict__ qb, const _Float16* __restrict__ kb,
                            const _Float16* __restrict__ vb, const float* __restrict__ bias,
                            _Float16* __restrict__ ao) {
  __shared__ __align__(16) char smem[49152];
  float*    S  = (float*)smem;                 // [2][NQ][NK] f32 = 32 KB
  _Float16* P  = (_Float16*)(smem + 32768);    // [2][NQ][NK] f16 = 16 KB
  _Float16* Vt = (_Float16*)smem;              // [64][NK] f16 = 16 KB (phase 3+)

  int blk = blockIdx.x;
  int hp = blk & 1;                    // head pair 0/1
  int wi = (blk >> 1) % NW;            // window
  int b = (blk >> 1) / NW;
  int lane = threadIdx.x & 31;
  int wv = threadIdx.x >> 5;
  int h_local = wv >> 1, mt = wv & 1;
  int hglob = hp * 2 + h_local;
  const size_t base = (size_t)b * N_ATOMS * C_ATOM;
  int q0 = wi * NQ + mt * 16;
  int kbase = wi * NQ - PAD;

  // ---- raw scores: S = Q Kt (no epilogue work in fragment store) ----
  v16h aq = load_fragA_f16(qb + base, C_ATOM, q0, hglob * 32, lane);
#pragma unroll
  for (int nt = 0; nt < 8; ++nt) {
    v16h bk = load_fragB_kclamp(kb + base, kbase + nt * 16, hglob * 32, lane);
    v8f sc = wmma16(aq, bk, v8f_zero());
    int kcol = nt * 16 + (lane & 15);
    int qw0 = mt * 16 + ((lane >> 4) << 3);
#pragma unroll
    for (int vv = 0; vv < 8; ++vv)
      S[(h_local * NQ + qw0 + vv) * NK + kcol] = sc[vv];
  }
  __syncthreads();

  // ---- softmax over k (row per thread, 64 rows); scale+bias+mask fused ----
  if (threadIdx.x < 2 * NQ) {
    int hl = threadIdx.x >> 5, qq = threadIdx.x & 31;
    float* r = S + (hl * NQ + qq) * NK;
    const float* brow = bias + (((size_t)b * NHEADS + hp * 2 + hl) * NQ + qq) * NK;
    _Float16* pr = P + (hl * NQ + qq) * NK;
    float m = -1e30f;
    for (int k = 0; k < NK; ++k) {
      bool ok = (unsigned)(kbase + k) < (unsigned)N_ATOMS;
      float val = ok ? (r[k] * SCALE_ATTN + brow[k]) : -1e9f;
      r[k] = val;
      m = fmaxf(m, val);
    }
    float sum = 0.f;
    for (int k = 0; k < NK; ++k) {
      float e = __expf(r[k] - m);
      r[k] = e; sum += e;
    }
    float inv = 1.f / sum;
    for (int k = 0; k < NK; ++k) pr[k] = (_Float16)(r[k] * inv);
  }
  __syncthreads();

  // ---- stage V window transposed into LDS (head pair: 64 dims x 128 keys) --
  {
    int t = threadIdx.x;                      // one key per thread
    int krow = kbase + t;
    krow = krow < 0 ? 0 : (krow > N_ATOMS - 1 ? N_ATOMS - 1 : krow);
    const _Float16* vrow = vb + base + (size_t)krow * C_ATOM + hp * 64;
#pragma unroll
    for (int d0 = 0; d0 < 64; d0 += 8) {
      v8h chunk = *(const v8h*)(vrow + d0);
#pragma unroll
      for (int j = 0; j < 8; ++j) Vt[(d0 + j) * NK + t] = chunk[j];
    }
  }
  __syncthreads();

  // ---- output: O = P @ V (A from LDS P, B from LDS Vt) ----
  v8f oacc[2] = {v8f_zero(), v8f_zero()};
#pragma unroll
  for (int kt = 0; kt < 4; ++kt) {
    v16h ap = load_fragA_f16(P + h_local * NQ * NK, NK, mt * 16, kt * 32, lane);
#pragma unroll
    for (int dt = 0; dt < 2; ++dt) {
      // B frag: lane n = local dim, contiguous keys from Vt row
      const _Float16* pB = Vt + (size_t)(h_local * 32 + dt * 16 + (lane & 15)) * NK +
                           kt * 32 + ((lane >> 4) << 4);
      v16h bv;
#pragma unroll
      for (int e = 0; e < 16; ++e) bv[e] = pB[e];
      oacc[dt] = wmma16(ap, bv, oacc[dt]);
    }
  }
#pragma unroll
  for (int dt = 0; dt < 2; ++dt) {
#pragma unroll
    for (int vv = 0; vv < 8; ++vv) {
      int qwin = mt * 16 + ((lane >> 4) << 3) + vv;
      ao[base + (size_t)(wi * NQ + qwin) * C_ATOM + hglob * 32 + dt * 16 + (lane & 15)] =
          (_Float16)oacc[dt][vv];
    }
  }
}

// ---------------------------------------------------------------------------
// Fused FFN: x += relu(h2 @ Wt1) @ Wt2. block = 128 thr (4 waves), 16 rows.
// mid [16,512] f16 staged in LDS between the two WMMA GEMMs.
// ---------------------------------------------------------------------------
__global__ void k_ffn_fused(const _Float16* __restrict__ h2,
                            const _Float16* __restrict__ Wt1T,
                            const _Float16* __restrict__ Wt2T,
                            float* __restrict__ x) {
  __shared__ __align__(16) _Float16 mid[16][512];    // 16 KB
  int lane = threadIdx.x & 31;
  int wv = threadIdx.x >> 5;
  int m0 = blockIdx.x * 16;

  v8f acc[8] = {v8f_zero(), v8f_zero(), v8f_zero(), v8f_zero(),
                v8f_zero(), v8f_zero(), v8f_zero(), v8f_zero()};
  for (int k0 = 0; k0 < 128; k0 += 32) {
    v16h a = load_fragA_f16(h2, C_ATOM, m0, k0, lane);
#pragma unroll
    for (int nt = 0; nt < 8; ++nt) {
      v16h b = load_fragB_f16(Wt1T, 128, wv * 128 + nt * 16, k0, lane);
      acc[nt] = wmma16(a, b, acc[nt]);
    }
  }
  int mb = (lane >> 4) << 3;
#pragma unroll
  for (int nt = 0; nt < 8; ++nt) {
    int n = wv * 128 + nt * 16 + (lane & 15);
#pragma unroll
    for (int vv = 0; vv < 8; ++vv)
      mid[mb + vv][n] = (_Float16)fmaxf(acc[nt][vv], 0.f);
  }
  __syncthreads();

  v8f oacc[2] = {v8f_zero(), v8f_zero()};
  for (int k0 = 0; k0 < 512; k0 += 32) {
    v16h a = load_fragA_f16(&mid[0][0], 512, 0, k0, lane);
#pragma unroll
    for (int nt = 0; nt < 2; ++nt) {
      v16h b = load_fragB_f16(Wt2T, 512, wv * 32 + nt * 16, k0, lane);
      oacc[nt] = wmma16(a, b, oacc[nt]);
    }
  }
#pragma unroll
  for (int nt = 0; nt < 2; ++nt) {
    int n = wv * 32 + nt * 16 + (lane & 15);
#pragma unroll
    for (int vv = 0; vv < 8; ++vv)
      x[(size_t)(m0 + mb + vv) * C_ATOM + n] += oacc[nt][vv];
  }
}

// ---------------------------------------------------------------------------
// Host orchestration
// ---------------------------------------------------------------------------
extern "C" void kernel_launch(void* const* d_in, const int* in_sizes, int n_in,
                              void* d_out, int out_size, void* d_ws, size_t ws_size,
                              hipStream_t stream) {
  const float* a      = (const float*)d_in[0];
  const int*   idx    = (const int*)d_in[2];
  const float* W_a    = (const float*)d_in[3];
  const float* W_out  = (const float*)d_in[4];
  const float* W_cl   = (const float*)d_in[5];
  const float* W_cm   = (const float*)d_in[6];
  const float* W_mlp1 = (const float*)d_in[7];
  const float* W_mlp2 = (const float*)d_in[8];
  const float* W_pb   = (const float*)d_in[9];
  const float* Wq     = (const float*)d_in[10];
  const float* Wk     = (const float*)d_in[11];
  const float* Wv     = (const float*)d_in[12];
  const float* Wo     = (const float*)d_in[13];
  const float* ln1_g  = (const float*)d_in[14];
  const float* ln1_b  = (const float*)d_in[15];
  const float* Wt1    = (const float*)d_in[16];
  const float* Wt2    = (const float*)d_in[17];
  const float* ln2_g  = (const float*)d_in[18];
  const float* ln2_b  = (const float*)d_in[19];

  const size_t MROWS = (size_t)BATCH * N_ATOMS;   // 65536
  char* wp = (char*)d_ws;
  auto alloc = [&](size_t bytes) -> void* {
    void* p = (void*)wp;
    wp += (bytes + 255) & ~(size_t)255;
    return p;
  };
  _Float16* W_aT   = (_Float16*)alloc(sizeof(_Float16) * 128 * 384);
  _Float16* W_outT = (_Float16*)alloc(sizeof(_Float16) * 128 * 128);
  _Float16* WqT    = (_Float16*)alloc(sizeof(_Float16) * 3 * 128 * 128);
  _Float16* WkT    = (_Float16*)alloc(sizeof(_Float16) * 3 * 128 * 128);
  _Float16* WvT    = (_Float16*)alloc(sizeof(_Float16) * 3 * 128 * 128);
  _Float16* WoT    = (_Float16*)alloc(sizeof(_Float16) * 3 * 128 * 128);
  _Float16* Wt1T   = (_Float16*)alloc(sizeof(_Float16) * 3 * 512 * 128);
  _Float16* Wt2T   = (_Float16*)alloc(sizeof(_Float16) * 3 * 128 * 512);
  float*    a_tok  = (float*)alloc(sizeof(float) * BATCH * N_TOK * C_ATOM);
  float*    x      = (float*)alloc(sizeof(float) * MROWS * C_ATOM);
  _Float16* hbuf   = (_Float16*)alloc(sizeof(_Float16) * MROWS * C_ATOM);
  _Float16* qbuf   = (_Float16*)alloc(sizeof(_Float16) * MROWS * C_ATOM);
  _Float16* kbuf   = (_Float16*)alloc(sizeof(_Float16) * MROWS * C_ATOM);
  _Float16* vbuf   = (_Float16*)alloc(sizeof(_Float16) * MROWS * C_ATOM);
  _Float16* aobuf  = (_Float16*)alloc(sizeof(_Float16) * MROWS * C_ATOM);
  float*    bias   = (float*)alloc(sizeof(float) * BATCH * NHEADS * NQ * NK);

  auto tpose = [&](const float* src, _Float16* dst, int K, int N) {
    int total = K * N;
    k_transpose_f32_f16<<<(total + 255) / 256, 256, 0, stream>>>(src, dst, K, N);
  };
  tpose(W_a, W_aT, 384, 128);
  tpose(W_out, W_outT, 128, 128);
  for (int l = 0; l < 3; ++l) {
    tpose(Wq + (size_t)l * 128 * 128, WqT + (size_t)l * 128 * 128, 128, 128);
    tpose(Wk + (size_t)l * 128 * 128, WkT + (size_t)l * 128 * 128, 128, 128);
    tpose(Wv + (size_t)l * 128 * 128, WvT + (size_t)l * 128 * 128, 128, 128);
    tpose(Wo + (size_t)l * 128 * 128, WoT + (size_t)l * 128 * 128, 128, 128);
    tpose(Wt1 + (size_t)l * 128 * 512, Wt1T + (size_t)l * 512 * 128, 128, 512);
    tpose(Wt2 + (size_t)l * 512 * 128, Wt2T + (size_t)l * 128 * 512, 512, 128);
  }

  // 1) token projection: a_tok = a @ W_a   (M=8192, K=384, N=128)
  {
    dim3 g(BATCH * N_TOK / 64, 128 / 64);
    k_gemm_wmma<true, 0, 384><<<g, 128, 0, stream>>>((const void*)a, W_aT,
                                                     (void*)a_tok, BATCH * N_TOK,
                                                     128, 384);
  }
  // 2) gather tokens -> atoms
  k_gather<<<(int)MROWS, 128, 0, stream>>>(a_tok, idx, x);
  // 3) pair embedding + attention bias (shared across windows)
  k_pair_bias<<<BATCH, 128, 0, stream>>>(x, W_cl, W_cm, W_mlp1, W_mlp2, W_pb, bias);

  dim3 gProj((unsigned)(MROWS / 64), 128 / 64);
  int lnBlocks = (int)(MROWS / 8);
  for (int l = 0; l < 3; ++l) {
    const _Float16* wqT = WqT + (size_t)l * 128 * 128;
    const _Float16* wkT = WkT + (size_t)l * 128 * 128;
    const _Float16* wvT = WvT + (size_t)l * 128 * 128;
    const _Float16* woT = WoT + (size_t)l * 128 * 128;
    // LN1 -> h (f16)
    k_layernorm_f16<<<lnBlocks, 256, 0, stream>>>(x, ln1_g + l * 128, ln1_b + l * 128, hbuf);
    // Q, K, V projections (f16 out)
    k_gemm_wmma<false, 2, 128><<<gProj, 128, 0, stream>>>((const void*)hbuf, wqT,
                                                          (void*)qbuf, (int)MROWS, 128, 128);
    k_gemm_wmma<false, 2, 128><<<gProj, 128, 0, stream>>>((const void*)hbuf, wkT,
                                                          (void*)kbuf, (int)MROWS, 128, 128);
    k_gemm_wmma<false, 2, 128><<<gProj, 128, 0, stream>>>((const void*)hbuf, wvT,
                                                          (void*)vbuf, (int)MROWS, 128, 128);
    // windowed attention
    k_attention<<<BATCH * NW * 2, 128, 0, stream>>>(qbuf, kbuf, vbuf, bias, aobuf);
    // x += ao @ Wo
    k_gemm_wmma<false, 1, 128><<<gProj, 128, 0, stream>>>((const void*)aobuf, woT,
                                                          (void*)x, (int)MROWS, 128, 128);
    // LN2 -> h2 (reuse hbuf)
    k_layernorm_f16<<<lnBlocks, 256, 0, stream>>>(x, ln2_g + l * 128, ln2_b + l * 128, hbuf);
    // x += relu(h2 @ Wt1) @ Wt2  (fused)
    k_ffn_fused<<<(int)(MROWS / 16), 128, 0, stream>>>(hbuf,
                                                       Wt1T + (size_t)l * 512 * 128,
                                                       Wt2T + (size_t)l * 128 * 512, x);
  }
  // final: out = x @ W_out
  k_gemm_wmma<true, 0, 128><<<gProj, 128, 0, stream>>>((const void*)x, W_outT, d_out,
                                                       (int)MROWS, 128, 128);
}
